// SlidingWindowAttention_11742440587390
// MI455X (gfx1250) — matrix operand order repaired and checked by reading, not compiled, not run on hardware
//
#include <hip/hip_runtime.h>

#define SEQ_LEN 4096
#define EMBED   2048
#define NHEADS  16
#define NKV     4
#define HDIM    128
#define KVREP   (NHEADS / NKV)
#define WIN     512

typedef __attribute__((ext_vector_type(8)))  _Float16 v8h;
typedef __attribute__((ext_vector_type(16))) _Float16 v16h;
typedef __attribute__((ext_vector_type(8)))  float    v8f;

static __device__ __forceinline__ v16h cat8(v8h a, v8h b) {
  return __builtin_shufflevector(a, b, 0,1,2,3,4,5,6,7,8,9,10,11,12,13,14,15);
}

// A fragment 16x32 (MxK), row-major source, stride lda, tile origin (row0,k0).
// lane m in [0,16): K = 0..15; lane m+16: K = 16..31 (16 contiguous halfs).
static __device__ __forceinline__ v16h load_a_frag(const _Float16* base, int lda,
                                                   int row0, int k0, int lane) {
  int m = lane & 15, hi = lane >> 4;
  const _Float16* p = base + (size_t)(row0 + m) * lda + k0 + hi * 16;
  v8h lo = *(const v8h*)p;
  v8h hh = *(const v8h*)(p + 8);
  return cat8(lo, hh);
}

// B fragment 32x16 (KxN) with B[k][n] = W[n0+n][k0+k], W row-major stride ldw.
// lane n in [0,16): K = 0..15; lane n+16: K = 16..31 (16 contiguous halfs).
static __device__ __forceinline__ v16h load_b_frag(const _Float16* W, int ldw,
                                                   int n0, int k0, int lane) {
  int n = lane & 15, hi = lane >> 4;
  const _Float16* p = W + (size_t)(n0 + n) * ldw + k0 + hi * 16;
  v8h lo = *(const v8h*)p;
  v8h hh = *(const v8h*)(p + 8);
  return cat8(lo, hh);
}

static __device__ __forceinline__ v8f wmma_f16(v16h a, v16h b, v8f c) {
  return __builtin_amdgcn_wmma_f32_16x16x32_f16(false, a, false, b, (short)0, c,
                                                false, false);
}

__global__ void cvt_f32_f16(const float* __restrict__ in, _Float16* __restrict__ out,
                            long n) {
  long i = (long)blockIdx.x * blockDim.x + threadIdx.x;
  long stride = (long)gridDim.x * blockDim.x;
  for (; i < n; i += stride) out[i] = (_Float16)in[i];
}

// y[M,N] = A[M,K] @ W[N,K]^T + bias; one wave computes a 32x64 tile
// (two A fragments amortize every B fragment: 12 loads per 8 WMMAs).
// mode 0: f16 row-major (ldo); mode 1: f16 transposed (out[col*ldo+row]);
// mode 2: f32 row-major (ldo).
__global__ void gemm_wmma(const _Float16* __restrict__ A, int lda,
                          const _Float16* __restrict__ W, int ldw,
                          const float* __restrict__ bias, int K,
                          _Float16* __restrict__ outH, float* __restrict__ outF,
                          int ldo, int mode) {
  int lane = threadIdx.x;
  int m0 = blockIdx.y * 32;
  int n0 = blockIdx.x * 64;
  v8f acc0[4] = {};
  v8f acc1[4] = {};
  for (int k0 = 0; k0 < K; k0 += 32) {
    v16h a0 = load_a_frag(A, lda, m0,      k0, lane);
    v16h a1 = load_a_frag(A, lda, m0 + 16, k0, lane);
#pragma unroll
    for (int t = 0; t < 4; ++t) {
      v16h b = load_b_frag(W, ldw, n0 + t * 16, k0, lane);
      acc0[t] = wmma_f16(a0, b, acc0[t]);
      acc1[t] = wmma_f16(a1, b, acc1[t]);
    }
  }
  int nn = lane & 15, hi = lane >> 4;
#pragma unroll
  for (int mi = 0; mi < 2; ++mi) {
#pragma unroll
    for (int t = 0; t < 4; ++t) {
      int col = n0 + t * 16 + nn;
      float bv = bias ? bias[col] : 0.0f;
#pragma unroll
      for (int r = 0; r < 8; ++r) {
        int row = m0 + mi * 16 + r + hi * 8;
        float v = (mi == 0 ? acc0[t][r] : acc1[t][r]) + bv;
        if (mode == 0)      outH[(size_t)row * ldo + col] = (_Float16)v;
        else if (mode == 1) outH[(size_t)col * ldo + row] = (_Float16)v;
        else                outF[(size_t)row * ldo + col] = v;
      }
    }
  }
}

// Flash-style causal sliding-window attention. One wave per (head, 16-query tile).
// Qh: [SEQ, EMBED] f16 (head h at cols h*128..); Kh: [SEQ, NKV*HDIM] f16;
// Vt: [NKV*HDIM, SEQ] f16 (transposed); ctx out: [SEQ, EMBED] f16.
__global__ void attn_swa(const _Float16* __restrict__ Qh,
                         const _Float16* __restrict__ Kh,
                         const _Float16* __restrict__ Vt,
                         _Float16* __restrict__ ctx) {
  __shared__ __align__(16) _Float16 lds_p[16 * 32];
  int lane = threadIdx.x;
  int qt = blockIdx.x & (SEQ_LEN / 16 - 1);
  int h  = blockIdx.x >> 8;                 // 256 q-tiles per head
  int kvh = h / KVREP;
  int q0 = qt * 16;
  int nn = lane & 15, hi = lane >> 4;
  const float scale = 0.08838834764831845f; // 1/sqrt(128)

  v16h qf[4];
#pragma unroll
  for (int c = 0; c < 4; ++c)
    qf[c] = load_a_frag(Qh, EMBED, q0, h * HDIM + c * 32, lane);

  v8f acc[8] = {};                           // 16 x 128 ctx accumulator (f32)
  float mrow[8], lrow[8];
#pragma unroll
  for (int r = 0; r < 8; ++r) { mrow[r] = -1e30f; lrow[r] = 0.0f; }

  int kstart = q0 - WIN; if (kstart < 0) kstart = 0;
  kstart &= ~31;
  int kend = q0 + 16;                        // exclusive
  for (int kb = kstart; kb < kend; kb += 32) {
    // ---- scores: two 16(q) x 16(key) tiles, K-dim = head_dim 128 ----
    v8f s0 = {}, s1 = {};
#pragma unroll
    for (int c = 0; c < 4; ++c) {
      int dk = kvh * HDIM + c * 32;
      s0 = wmma_f16(qf[c], load_b_frag(Kh, NKV * HDIM, kb,      dk, lane), s0);
      s1 = wmma_f16(qf[c], load_b_frag(Kh, NKV * HDIM, kb + 16, dk, lane), s1);
    }
    // ---- mask + online softmax (rows live per-element within a 16-lane half) ----
#pragma unroll
    for (int r = 0; r < 8; ++r) {
      int qrow = q0 + r + hi * 8;
      int k0c = kb + nn, k1c = kb + 16 + nn;
      float v0 = s0[r] * scale;
      float v1 = s1[r] * scale;
      if (k0c > qrow || k0c < qrow - WIN) v0 = -1e30f;
      if (k1c > qrow || k1c < qrow - WIN) v1 = -1e30f;
      float mx = fmaxf(v0, v1);
      mx = fmaxf(mx, __shfl_xor(mx, 1, 16));
      mx = fmaxf(mx, __shfl_xor(mx, 2, 16));
      mx = fmaxf(mx, __shfl_xor(mx, 4, 16));
      mx = fmaxf(mx, __shfl_xor(mx, 8, 16));
      float mnew = fmaxf(mrow[r], mx);
      float sc = __expf(mrow[r] - mnew);
      float p0 = (v0 < -1e29f) ? 0.0f : __expf(v0 - mnew);
      float p1 = (v1 < -1e29f) ? 0.0f : __expf(v1 - mnew);
      float rs = p0 + p1;
      rs += __shfl_xor(rs, 1, 16);
      rs += __shfl_xor(rs, 2, 16);
      rs += __shfl_xor(rs, 4, 16);
      rs += __shfl_xor(rs, 8, 16);
      lrow[r] = lrow[r] * sc + rs;
      mrow[r] = mnew;
#pragma unroll
      for (int dt = 0; dt < 8; ++dt) acc[dt][r] *= sc;
      int row = r + hi * 8;
      lds_p[row * 32 + nn]      = (_Float16)p0;
      lds_p[row * 32 + 16 + nn] = (_Float16)p1;
    }
    __syncthreads();
    // ---- P (A-layout from LDS) @ V (B from transposed Vt) ----
    {
      const _Float16* pp = lds_p + nn * 32 + hi * 16;
      v8h l0 = *(const v8h*)pp;
      v8h l1 = *(const v8h*)(pp + 8);
      v16h pf = cat8(l0, l1);
#pragma unroll
      for (int dt = 0; dt < 8; ++dt) {
        const _Float16* vp = Vt + (size_t)(kvh * HDIM + dt * 16 + nn) * SEQ_LEN
                                + kb + hi * 16;
        v8h b0 = *(const v8h*)vp;
        v8h b1 = *(const v8h*)(vp + 8);
        acc[dt] = wmma_f16(pf, cat8(b0, b1), acc[dt]);
      }
    }
    __syncthreads();
  }
  // ---- normalize + store ctx (f16) ----
#pragma unroll
  for (int dt = 0; dt < 8; ++dt) {
    int col = h * HDIM + dt * 16 + nn;
#pragma unroll
    for (int r = 0; r < 8; ++r) {
      int row = q0 + r + hi * 8;
      float v = acc[dt][r] / lrow[r];
      ctx[(size_t)row * EMBED + col] = (_Float16)v;
    }
  }
}

extern "C" void kernel_launch(void* const* d_in, const int* in_sizes, int n_in,
                              void* d_out, int out_size, void* d_ws, size_t ws_size,
                              hipStream_t stream) {
  const float* x    = (const float*)d_in[0];
  const float* wq_w = (const float*)d_in[1];
  const float* wq_b = (const float*)d_in[2];
  const float* wk_w = (const float*)d_in[3];
  const float* wk_b = (const float*)d_in[4];
  const float* wv_w = (const float*)d_in[5];
  const float* wv_b = (const float*)d_in[6];
  const float* wo_w = (const float*)d_in[7];
  const float* wo_b = (const float*)d_in[8];
  float* out = (float*)d_out;

  char* ws = (char*)d_ws;
  size_t off = 0;
  auto carve = [&](size_t bytes) -> char* {
    char* p = ws + off;
    off += (bytes + 255) & ~(size_t)255;
    return p;
  };
  _Float16* xh  = (_Float16*)carve((size_t)SEQ_LEN * EMBED * 2);          // 16 MB
  _Float16* wqh = (_Float16*)carve((size_t)EMBED * EMBED * 2);            //  8 MB
  _Float16* wkh = (_Float16*)carve((size_t)NKV * HDIM * EMBED * 2);       //  2 MB
  _Float16* wvh = (_Float16*)carve((size_t)NKV * HDIM * EMBED * 2);       //  2 MB
  _Float16* woh = (_Float16*)carve((size_t)EMBED * EMBED * 2);            //  8 MB
  _Float16* Qh  = (_Float16*)carve((size_t)SEQ_LEN * EMBED * 2);          // 16 MB
  _Float16* Kh  = (_Float16*)carve((size_t)SEQ_LEN * NKV * HDIM * 2);     //  4 MB
  _Float16* Vt  = (_Float16*)carve((size_t)NKV * HDIM * SEQ_LEN * 2);     //  4 MB
  _Float16* ctx = (_Float16*)carve((size_t)SEQ_LEN * EMBED * 2);          // 16 MB
  (void)ws_size; (void)in_sizes; (void)n_in; (void)out_size;

  dim3 cb(256);
  cvt_f32_f16<<<dim3(2048), cb, 0, stream>>>(x,    xh,  (long)SEQ_LEN * EMBED);
  cvt_f32_f16<<<dim3(1024), cb, 0, stream>>>(wq_w, wqh, (long)EMBED * EMBED);
  cvt_f32_f16<<<dim3(512),  cb, 0, stream>>>(wk_w, wkh, (long)NKV * HDIM * EMBED);
  cvt_f32_f16<<<dim3(512),  cb, 0, stream>>>(wv_w, wvh, (long)NKV * HDIM * EMBED);
  cvt_f32_f16<<<dim3(1024), cb, 0, stream>>>(wo_w, woh, (long)EMBED * EMBED);

  dim3 wave(32);
  // Q = x @ wq^T + b  -> f16 [SEQ, EMBED]
  gemm_wmma<<<dim3(EMBED / 64, SEQ_LEN / 32), wave, 0, stream>>>(
      xh, EMBED, wqh, EMBED, wq_b, EMBED, Qh, nullptr, EMBED, 0);
  // K = x @ wk^T + b  -> f16 [SEQ, 512] row-major (B-operand of Q.K^T)
  gemm_wmma<<<dim3((NKV * HDIM) / 64, SEQ_LEN / 32), wave, 0, stream>>>(
      xh, EMBED, wkh, EMBED, wk_b, EMBED, Kh, nullptr, NKV * HDIM, 0);
  // V = x @ wv^T + b  -> f16 transposed [512, SEQ] (B-operand of P.V)
  gemm_wmma<<<dim3((NKV * HDIM) / 64, SEQ_LEN / 32), wave, 0, stream>>>(
      xh, EMBED, wvh, EMBED, wv_b, EMBED, Vt, nullptr, SEQ_LEN, 1);

  // sliding-window attention: 16 heads x 256 query tiles
  attn_swa<<<dim3(NHEADS * (SEQ_LEN / 16)), wave, 0, stream>>>(Qh, Kh, Vt, ctx);

  // out = ctx @ wo^T + b -> f32 d_out
  gemm_wmma<<<dim3(EMBED / 64, SEQ_LEN / 32), wave, 0, stream>>>(
      ctx, EMBED, woh, EMBED, wo_b, EMBED, nullptr, out, EMBED, 2);
}
